// RadianceFieldTransformer_67645734912499
// MI455X (gfx1250) — matrix operand order repaired and checked by reading, not compiled, not run on hardware
//
#include <hip/hip_runtime.h>

typedef _Float16 half_t;
typedef __attribute__((ext_vector_type(16))) _Float16 v16h;
typedef __attribute__((ext_vector_type(8)))  float    v8f;

#define EMBED   256
#define HIDDEN  1024
#define HEADS   8
#define LSEQ    1024
#define MTOK    2048   // B * L
#define GB_N    64     // GEMM block tile N
#define GB_M    64     // GEMM block tile M (4 waves x 16 rows)

// ---------------------------------------------------------------------------
// WMMA + async helpers
// ---------------------------------------------------------------------------
__device__ __forceinline__ v8f wmma_f16(v16h a, v16h b, v8f c) {
  return __builtin_amdgcn_wmma_f32_16x16x32_f16(
      /*neg_a=*/false, a, /*neg_b=*/false, b,
      /*c_mod=*/(short)0, c, /*reuse_a=*/false, /*reuse_b=*/false);
}

// Per-lane 16-byte async copy global -> LDS (ASYNCcnt-tracked, no VGPR data).
__device__ __forceinline__ void async_load_b128(unsigned lds_addr,
                                                const void* gaddr) {
  asm volatile("global_load_async_to_lds_b128 %0, %1, off"
               :: "v"(lds_addr), "v"(gaddr) : "memory");
}
__device__ __forceinline__ void wait_async0() {
#if __has_builtin(__builtin_amdgcn_s_wait_asynccnt)
  __builtin_amdgcn_s_wait_asynccnt(0);
#else
  asm volatile("s_wait_asynccnt 0x0" ::: "memory");
#endif
}
// Low 32 bits of a generic pointer into LDS == raw LDS byte offset.
__device__ __forceinline__ unsigned lds_off(const void* p) {
  return (unsigned)(unsigned long long)p;
}

// Load a 16x32 f16 A-fragment (or 32x16 B-fragment from an [N,K] transposed
// matrix): lane L<16 -> row/col (row0+L), elems k..k+15; lanes 16-31 -> +16 k.
__device__ __forceinline__ v16h ldfrag(const half_t* __restrict__ base,
                                       int row0, int ld, int kbase) {
  int lane = threadIdx.x & 31;
  const half_t* p = base + (size_t)(row0 + (lane & 15)) * ld
                         + kbase + ((lane >> 4) << 4);
  return *(const v16h*)p;
}

__device__ __forceinline__ float hmax16(float v) {
  for (int m = 1; m < 16; m <<= 1) v = fmaxf(v, __shfl_xor(v, m, 16));
  return v;
}
__device__ __forceinline__ float hsum16(float v) {
  for (int m = 1; m < 16; m <<= 1) v += __shfl_xor(v, m, 16);
  return v;
}

// ---------------------------------------------------------------------------
// WMMA GEMM: C[M,N] = A[M,K] (f16 row-major) @ BT[N,K] (f16).
// Block = 128 threads (4 waves) computing a 64x64 macro-tile; wave w owns a
// 16x64 strip (4 accumulators, A-frag reused 4x). The shared 64x32 B tile is
// staged to LDS with global_load_async_to_lds_b128, double-buffered.
// Epilogue: +bias, +residual(f32), ReLU; f32 / f16 / per-head-transposed f16.
// grid = (N/64, M/64).
// ---------------------------------------------------------------------------
__global__ __launch_bounds__(128) void gemm_wmma(
    const half_t* __restrict__ A, const half_t* __restrict__ BT,
    const float* __restrict__ bias, const float* __restrict__ resid,
    float* __restrict__ Cf, half_t* __restrict__ Ch, half_t* __restrict__ ChT,
    int M, int N, int K, int Lseq, int relu) {
  __shared__ __align__(64) _Float16 Bs[2][GB_N * 32];
  int tid  = threadIdx.x;
  int lane = tid & 31;
  int wave = tid >> 5;
  int n0 = blockIdx.x * GB_N;
  int m0 = blockIdx.y * GB_M + wave * 16;

  // This thread's 32-byte slice of the B tile: row n0+brow, halves bcol..+15.
  int brow = tid >> 1;
  int bcol = (tid & 1) << 4;
  const half_t* bsrc = BT + (size_t)(n0 + brow) * K + bcol;
  unsigned bdst0 = lds_off(&Bs[0][brow * 32 + bcol]);
  unsigned bdst1 = lds_off(&Bs[1][brow * 32 + bcol]);

  async_load_b128(bdst0, bsrc);  // stage k=0 into buffer 0
  async_load_b128(bdst0 + 16, bsrc + 8);

  v8f acc0 = {}, acc1 = {}, acc2 = {}, acc3 = {};
  int frag_off = ((lane & 15) * 32) + ((lane >> 4) << 4);
  int buf = 0;
  for (int k = 0; k < K; k += 32) {
    wait_async0();
    __syncthreads();  // everyone's staging for `buf` has landed
    if (k + 32 < K) {
      async_load_b128(buf ? bdst0 : bdst1, bsrc + k + 32);
      async_load_b128((buf ? bdst0 : bdst1) + 16, bsrc + k + 40);
      __builtin_prefetch(A + (size_t)m0 * K + k + 32, 0, 0);
    }
    v16h a = ldfrag(A, m0, K, k);
    const _Float16* bb = &Bs[buf][0];
    v16h b0 = *(const v16h*)&bb[frag_off];
    v16h b1 = *(const v16h*)&bb[frag_off + 16 * 32];
    v16h b2 = *(const v16h*)&bb[frag_off + 32 * 32];
    v16h b3 = *(const v16h*)&bb[frag_off + 48 * 32];
    acc0 = wmma_f16(a, b0, acc0);
    acc1 = wmma_f16(a, b1, acc1);
    acc2 = wmma_f16(a, b2, acc2);
    acc3 = wmma_f16(a, b3, acc3);
    buf ^= 1;
  }

  auto epilogue = [&](v8f ac, int j) {
    int n = n0 + (j << 4) + (lane & 15);
    float bi = bias ? bias[n] : 0.0f;
    for (int r = 0; r < 8; ++r) {
      int m = m0 + r + ((lane >> 4) << 3);
      float v = ac[r] + bi;
      if (resid) v += resid[(size_t)m * N + n];
      if (relu) v = fmaxf(v, 0.0f);
      if (Cf) Cf[(size_t)m * N + n] = v;
      if (Ch) Ch[(size_t)m * N + n] = (half_t)v;
      if (ChT) {  // V path: [B,H,32,L]
        int bb2 = m / Lseq, tok = m % Lseq;
        int hh = n >> 5, d = n & 31;
        ChT[((((size_t)bb2 * HEADS + hh) << 5) + d) * Lseq + tok] = (half_t)v;
      }
    }
  };
  epilogue(acc0, 0);
  epilogue(acc1, 1);
  epilogue(acc2, 2);
  epilogue(acc3, 3);
}

// ---------------------------------------------------------------------------
// Flash attention: one wave per 16-query tile of one (b,h), streaming K/V in
// 32-token steps with online softmax. mode=1: scores from Q@K^T WMMA;
// mode=0: scores[q,k] = s_q - s_k + sb[h] (pos-difference attention is
// linear, so the pairwise tensor collapses to a rank-1 difference).
// grid = (L/16, B*H), block = 32.
// ---------------------------------------------------------------------------
__global__ __launch_bounds__(32) void flash_attn(
    const half_t* __restrict__ Q16, const half_t* __restrict__ K16,
    const half_t* __restrict__ VT16,  // [B,H,32,L]
    const float* __restrict__ spos,   // [B*L, 8] (pos mode)
    const float* __restrict__ sbp,    // [8]     (pos mode)
    half_t* __restrict__ O16, int mode) {
  __shared__ __align__(64) _Float16 Pslab[16 * 32];
  int lane = threadIdx.x & 31;
  int nlo = lane & 15, half_id = lane >> 4;
  int q0 = blockIdx.x << 4;
  int bh = blockIdx.y;
  int b = bh >> 3, h = bh & 7;
  int rowbase = b * LSEQ;

  v16h aq = {};
  float sq[8];
  if (mode) {
    aq = ldfrag(Q16 + (size_t)h * 32, rowbase + q0, EMBED, 0);
  } else {
    float sbv = sbp[h];
    for (int r = 0; r < 8; ++r)
      sq[r] = spos[(size_t)(rowbase + q0 + r + (half_id << 3)) * HEADS + h] + sbv;
  }

  v8f acc0 = {}, acc1 = {};
  float mrow[8], lrow[8];
  for (int r = 0; r < 8; ++r) { mrow[r] = -1e30f; lrow[r] = 0.0f; }
  const float scale = 0.17677669529663687f;  // 1/sqrt(32)
  const half_t* vb = VT16 + ((size_t)bh << 5) * LSEQ;

  for (int k0 = 0; k0 < LSEQ; k0 += 32) {
    v8f slo = {}, shi = {};
    if (mode) {
      v16h bk0 = ldfrag(K16 + (size_t)h * 32, rowbase + k0, EMBED, 0);
      v16h bk1 = ldfrag(K16 + (size_t)h * 32, rowbase + k0 + 16, EMBED, 0);
      v8f z = {};
      slo = wmma_f16(aq, bk0, z);
      shi = wmma_f16(aq, bk1, z);
      for (int r = 0; r < 8; ++r) { slo[r] *= scale; shi[r] *= scale; }
    } else {
      float sklo = spos[(size_t)(rowbase + k0 + nlo) * HEADS + h];
      float skhi = spos[(size_t)(rowbase + k0 + 16 + nlo) * HEADS + h];
      for (int r = 0; r < 8; ++r) { slo[r] = sq[r] - sklo; shi[r] = sq[r] - skhi; }
    }
    // online softmax update (row stats replicated across each half-wave)
    for (int r = 0; r < 8; ++r) {
      float mn = fmaxf(mrow[r], hmax16(fmaxf(slo[r], shi[r])));
      float alpha = __expf(mrow[r] - mn);
      mrow[r] = mn;
      float plo = __expf(slo[r] - mn);
      float phi = __expf(shi[r] - mn);
      lrow[r] = lrow[r] * alpha + hsum16(plo + phi);
      acc0[r] *= alpha;
      acc1[r] *= alpha;
      int Mr = r + (half_id << 3);
      Pslab[Mr * 32 + nlo]      = (_Float16)plo;
      Pslab[Mr * 32 + 16 + nlo] = (_Float16)phi;
    }
    __syncthreads();
    // C-layout -> A-frag layout transpose via LDS, then P(16x32) @ V(32x16)x2
    v16h ap = *(const v16h*)&Pslab[nlo * 32 + (half_id << 4)];
    v16h bv0 = ldfrag(vb, 0, LSEQ, k0);
    v16h bv1 = ldfrag(vb, 16, LSEQ, k0);
    acc0 = wmma_f16(ap, bv0, acc0);
    acc1 = wmma_f16(ap, bv1, acc1);
    __syncthreads();
  }
  for (int r = 0; r < 8; ++r) {
    float inv = 1.0f / lrow[r];
    int m = rowbase + q0 + r + (half_id << 3);
    size_t o = (size_t)m * EMBED + h * 32;
    O16[o + nlo]      = (half_t)(acc0[r] * inv);
    O16[o + 16 + nlo] = (half_t)(acc1[r] * inv);
  }
}

// ---------------------------------------------------------------------------
// LayerNorm over the 256-wide feature dim, f16 output. grid = rows, block=256.
// ---------------------------------------------------------------------------
__global__ __launch_bounds__(256) void layernorm_cast(
    const float* __restrict__ x, const float* __restrict__ g,
    const float* __restrict__ be, half_t* __restrict__ out) {
  __shared__ float sdata[256];
  int row = blockIdx.x, tid = threadIdx.x;
  float v = x[(size_t)row * EMBED + tid];
  sdata[tid] = v;
  __syncthreads();
  for (int s = 128; s > 0; s >>= 1) {
    if (tid < s) sdata[tid] += sdata[tid + s];
    __syncthreads();
  }
  float mu = sdata[0] * (1.0f / EMBED);
  __syncthreads();
  float d = v - mu;
  sdata[tid] = d * d;
  __syncthreads();
  for (int s = 128; s > 0; s >>= 1) {
    if (tid < s) sdata[tid] += sdata[tid + s];
    __syncthreads();
  }
  float rs = rsqrtf(sdata[0] * (1.0f / EMBED) + 1e-6f);
  out[(size_t)row * EMBED + tid] = (half_t)(d * rs * g[tid] + be[tid]);
}

// ---------------------------------------------------------------------------
// Cast fp32 W[K,N] -> f16 WT[N,K]
// ---------------------------------------------------------------------------
__global__ void cast_transpose(const float* __restrict__ W,
                               half_t* __restrict__ WT, int K, int N) {
  int t = blockIdx.x * blockDim.x + threadIdx.x;
  if (t >= K * N) return;
  int k = t / N, n = t % N;
  WT[(size_t)n * K + k] = (half_t)W[(size_t)k * N + n];
}

// ---------------------------------------------------------------------------
// pos-mode score head: pe = relu(pos@pw1+pb1)@pw2+pb2 ; s[t,h] = pe @ sw[:,h]
// ---------------------------------------------------------------------------
__global__ __launch_bounds__(256) void pos_prep(
    const float* __restrict__ pos, const float* __restrict__ pw1,
    const float* __restrict__ pb1, const float* __restrict__ pw2,
    const float* __restrict__ pb2, const float* __restrict__ sw,
    float* __restrict__ spos) {
  int t = blockIdx.x * blockDim.x + threadIdx.x;
  if (t >= MTOK) return;
  float p4[4], hid[4], s[8];
  for (int c = 0; c < 4; ++c) p4[c] = pos[(size_t)t * 4 + c];
  for (int j = 0; j < 4; ++j) {
    float a = pb1[j];
    for (int c = 0; c < 4; ++c) a += p4[c] * pw1[c * 4 + j];
    hid[j] = fmaxf(a, 0.0f);
  }
  for (int h = 0; h < 8; ++h) s[h] = 0.0f;
  for (int o = 0; o < 32; ++o) {
    float pe = pb2[o];
    for (int j = 0; j < 4; ++j) pe += hid[j] * pw2[j * 32 + o];
    for (int h = 0; h < 8; ++h) s[h] += pe * sw[o * 8 + h];
  }
  for (int h = 0; h < 8; ++h) spos[(size_t)t * 8 + h] = s[h];
}

// ---------------------------------------------------------------------------
// Host orchestration
// ---------------------------------------------------------------------------
struct BP {
  const float *wv, *wo, *bo, *g1, *b1, *g2, *b2, *w1, *fb1, *w2, *fb2;
  const float *wq, *wk;
  const float *pw1, *pb1, *pw2, *pb2, *sw, *sb;
  int qk;
};

extern "C" void kernel_launch(void* const* d_in, const int* in_sizes, int n_in,
                              void* d_out, int out_size, void* d_ws, size_t ws_size,
                              hipStream_t stream) {
  const float* x0   = (const float*)d_in[0];
  const float* posi = (const float*)d_in[1];

  const int qkmode[4] = {0, 1, 0, 1};  // MODES = (pos, qk, pos, qk)
  BP bp[4];
  int idx = 2;
  for (int i = 0; i < 4; ++i) {
    BP& p = bp[i];
    p.qk  = qkmode[i];
    p.wv  = (const float*)d_in[idx++]; p.wo  = (const float*)d_in[idx++];
    p.bo  = (const float*)d_in[idx++]; p.g1  = (const float*)d_in[idx++];
    p.b1  = (const float*)d_in[idx++]; p.g2  = (const float*)d_in[idx++];
    p.b2  = (const float*)d_in[idx++]; p.w1  = (const float*)d_in[idx++];
    p.fb1 = (const float*)d_in[idx++]; p.w2  = (const float*)d_in[idx++];
    p.fb2 = (const float*)d_in[idx++];
    if (p.qk) {
      p.wq = (const float*)d_in[idx++]; p.wk = (const float*)d_in[idx++];
      p.pw1 = p.pb1 = p.pw2 = p.pb2 = p.sw = p.sb = nullptr;
    } else {
      p.pw1 = (const float*)d_in[idx++]; p.pb1 = (const float*)d_in[idx++];
      p.pw2 = (const float*)d_in[idx++]; p.pb2 = (const float*)d_in[idx++];
      p.sw  = (const float*)d_in[idx++]; p.sb  = (const float*)d_in[idx++];
      p.wq = p.wk = nullptr;
    }
  }

  // ---- workspace carve-out ----
  char* w = (char*)d_ws;
  auto alloc = [&](size_t bytes) -> void* {
    void* p = (void*)w;
    w += (bytes + 255) & ~(size_t)255;
    return p;
  };
  half_t* h16   = (half_t*)alloc((size_t)MTOK * EMBED * 2);
  half_t* q16   = (half_t*)alloc((size_t)MTOK * EMBED * 2);
  half_t* k16   = (half_t*)alloc((size_t)MTOK * EMBED * 2);
  half_t* vt16  = (half_t*)alloc((size_t)MTOK * EMBED * 2);   // [B,H,32,L]
  half_t* att16 = (half_t*)alloc((size_t)MTOK * EMBED * 2);
  half_t* t16   = (half_t*)alloc((size_t)MTOK * HIDDEN * 2);
  float*  bufA  = (float*)alloc((size_t)MTOK * EMBED * 4);
  float*  bufB  = (float*)alloc((size_t)MTOK * EMBED * 4);
  half_t* wqT   = (half_t*)alloc((size_t)EMBED * EMBED * 2);
  half_t* wkT   = (half_t*)alloc((size_t)EMBED * EMBED * 2);
  half_t* wvT   = (half_t*)alloc((size_t)EMBED * EMBED * 2);
  half_t* woT   = (half_t*)alloc((size_t)EMBED * EMBED * 2);
  half_t* w1T   = (half_t*)alloc((size_t)EMBED * HIDDEN * 2);
  half_t* w2T   = (half_t*)alloc((size_t)EMBED * HIDDEN * 2);
  float*  spos  = (float*)alloc((size_t)MTOK * HEADS * 4);

  const int cEE = (EMBED * EMBED + 255) / 256;
  const int cEH = (EMBED * HIDDEN + 255) / 256;
  dim3 gEE(EMBED / GB_N, MTOK / GB_M);    // 4 x 32 macro-tiles
  dim3 gEH(HIDDEN / GB_N, MTOK / GB_M);   // 16 x 32 macro-tiles
  dim3 gFA(LSEQ / 16, 2 * HEADS);         // 64 x 16

  float* bufs[2] = {bufA, bufB};
  const float* cur = x0;

  for (int i = 0; i < 4; ++i) {
    BP& p = bp[i];
    // --- attention half ---
    layernorm_cast<<<MTOK, 256, 0, stream>>>(cur, p.g1, p.b1, h16);
    cast_transpose<<<cEE, 256, 0, stream>>>(p.wv, wvT, EMBED, EMBED);
    gemm_wmma<<<gEE, 128, 0, stream>>>(h16, wvT, nullptr, nullptr,
                                       nullptr, nullptr, vt16,
                                       MTOK, EMBED, EMBED, LSEQ, 0);
    if (p.qk) {
      cast_transpose<<<cEE, 256, 0, stream>>>(p.wq, wqT, EMBED, EMBED);
      cast_transpose<<<cEE, 256, 0, stream>>>(p.wk, wkT, EMBED, EMBED);
      gemm_wmma<<<gEE, 128, 0, stream>>>(h16, wqT, nullptr, nullptr,
                                         nullptr, q16, nullptr,
                                         MTOK, EMBED, EMBED, LSEQ, 0);
      gemm_wmma<<<gEE, 128, 0, stream>>>(h16, wkT, nullptr, nullptr,
                                         nullptr, k16, nullptr,
                                         MTOK, EMBED, EMBED, LSEQ, 0);
    } else {
      pos_prep<<<MTOK / 256, 256, 0, stream>>>(posi, p.pw1, p.pb1, p.pw2,
                                               p.pb2, p.sw, spos);
    }
    flash_attn<<<gFA, 32, 0, stream>>>(q16, k16, vt16, spos, p.sb, att16, p.qk);

    cast_transpose<<<cEE, 256, 0, stream>>>(p.wo, woT, EMBED, EMBED);
    float* xa = bufs[i & 1];
    gemm_wmma<<<gEE, 128, 0, stream>>>(att16, woT, p.bo, cur,
                                       xa, nullptr, nullptr,
                                       MTOK, EMBED, EMBED, LSEQ, 0);
    // --- FFN half ---
    layernorm_cast<<<MTOK, 256, 0, stream>>>(xa, p.g2, p.b2, h16);
    cast_transpose<<<cEH, 256, 0, stream>>>(p.w1, w1T, EMBED, HIDDEN);
    gemm_wmma<<<gEH, 128, 0, stream>>>(h16, w1T, p.fb1, nullptr,
                                       nullptr, t16, nullptr,
                                       MTOK, HIDDEN, EMBED, LSEQ, 1);
    cast_transpose<<<cEH, 256, 0, stream>>>(p.w2, w2T, HIDDEN, EMBED);
    float* xb = (i == 3) ? (float*)d_out : bufs[(i + 1) & 1];
    gemm_wmma<<<gEE, 128, 0, stream>>>(t16, w2T, p.fb2, xa,
                                       xb, nullptr, nullptr,
                                       MTOK, EMBED, HIDDEN, LSEQ, 0);
    cur = xb;
  }
}
